// Encoder_26628797235385
// MI455X (gfx1250) — matrix operand order repaired and checked by reading, not compiled, not run on hardware
//
#include <hip/hip_runtime.h>
#include <hip/hip_bf16.h>

// ---------------- problem constants ----------------
#define B_     2
#define N_LL   65536
#define N_G    10242
#define F_IN   78
#define E_ENC  65536
#define E_LAT  (N_G * 7)        // 71694
#define N_TOT  (N_LL + N_G)     // 75778

// ---------------- tiling constants -----------------
#define TM       128            // rows per block
#define AP       264            // activation LDS pitch (halves)
#define WP       264            // weight LDS pitch (halves) == 256 + TDM pad (4 dwords)
#define THREADS  256            // 8 wave32
#define CHUNK_ELEMS (256 * 256) // one pre-converted bf16 weight chunk
#define SMEM_BYTES ((TM * AP + 256 * WP) * 2)   // 202,752 B dynamic LDS (<320KB/WGP)

typedef __attribute__((ext_vector_type(16))) __bf16 v16bf;
typedef __attribute__((ext_vector_type(8)))  __bf16 v8bf;
typedef __attribute__((ext_vector_type(8)))  float  v8f;
typedef __attribute__((ext_vector_type(4)))  unsigned int u32x4;
typedef __attribute__((ext_vector_type(8)))  int i32x8;
typedef __attribute__((ext_vector_type(4)))  int i32x4;

__device__ __forceinline__ unsigned short f2bf(float x) {
  union { float f; unsigned u; } c; c.f = x;
  unsigned u = c.u;
  u = u + 0x7FFFu + ((u >> 16) & 1u);       // round-to-nearest-even
  return (unsigned short)(u >> 16);
}

// two 16B LDS reads -> one 16-element bf16 fragment
__device__ __forceinline__ v16bf ld_frag(const unsigned short* p0, const unsigned short* p1) {
  v8bf lo = *(const v8bf*)p0;
  v8bf hi = *(const v8bf*)p1;
  return __builtin_shufflevector(lo, hi, 0,1,2,3,4,5,6,7,8,9,10,11,12,13,14,15);
}

// ---------------------------------------------------------------------------
// TDM: DMA one pre-converted bf16 weight chunk [256 rows x 256 halves] from
// global into LDS at byte offset ldsOff, inserting 4 DWORDs of pad per
// 128-DWORD row -> effective LDS pitch = 264 halves (== WP).
// ---------------------------------------------------------------------------
#if defined(__has_builtin)
#if __has_builtin(__builtin_amdgcn_tensor_load_to_lds) && __has_builtin(__builtin_amdgcn_s_wait_tensorcnt)
#define HAVE_TDM 1
#endif
#endif

__device__ __forceinline__ void tdm_load_chunk(const unsigned short* gsrc, unsigned ldsOff) {
#ifdef HAVE_TDM
  unsigned long long ga = (unsigned long long)gsrc;
  u32x4 g0;
  g0[0] = 1u;                                              // count=1 (user descriptor)
  g0[1] = ldsOff;                                          // lds_addr (bytes)
  g0[2] = (unsigned)ga;                                    // global_addr[31:0]
  g0[3] = (unsigned)((ga >> 32) & 0x01FFFFFFu) | (2u << 30); // global_addr[56:32] | type=2
  i32x8 g1;
  g1[0] = (1 << 16)        // data_size = 2 bytes
        | (1 << 20)        // pad_enable
        | (6 << 22)        // pad_interval: 128 DWORDs (one 256-half row)
        | (3 << 25);       // pad_amount: 4 DWORDs (8 halves)
  g1[1] = (int)(256u << 16);   // tensor_dim0[15:0]=256 (in bits 63:48 of group)
  g1[2] = (int)(256u << 16);   // tensor_dim0 hi=0 | tensor_dim1[15:0]=256
  g1[3] = (int)(256u << 16);   // tensor_dim1 hi=0 | tile_dim0=256
  g1[4] = 256;                 // tile_dim1=256, tile_dim2=0
  g1[5] = 256;                 // tensor_dim0_stride[31:0]=256 elements
  g1[6] = 0;                   // stride0 hi | tensor_dim1_stride lo (unused, 2D)
  g1[7] = 0;
  i32x4 gz = {0, 0, 0, 0};
#if __clang_major__ >= 23
  i32x8 gz8 = {0, 0, 0, 0, 0, 0, 0, 0};
  __builtin_amdgcn_tensor_load_to_lds(g0, g1, gz, gz, gz8, 0);
#else
  __builtin_amdgcn_tensor_load_to_lds(g0, g1, gz, gz, 0);
#endif
#else
  (void)gsrc; (void)ldsOff;
#endif
}

__device__ __forceinline__ void tdm_wait() {
#ifdef HAVE_TDM
  __builtin_amdgcn_s_wait_tensorcnt(0);
#endif
}

// ---------------------------------------------------------------------------
// Pre-pass: W[K][256] f32 -> bf16 transposed chunk dst[n*256 + k] (k local to
// [kc0, kc0+256), zero-padded beyond Kvalid). Grid 256 x 256 threads.
// ---------------------------------------------------------------------------
__global__ void wtconv_kernel(const float* __restrict__ W, int Kvalid, int kc0,
                              unsigned short* __restrict__ dst) {
  int t = blockIdx.x * 256 + threadIdx.x;   // 0..65535
  int n = t >> 8, k = t & 255;
  int kg = kc0 + k;
  float v = (kg < Kvalid) ? W[(size_t)kg * 256 + n] : 0.0f;
  dst[(size_t)n * 256 + k] = f2bf(v);
}

// ---------------------------------------------------------------------------
// Fused 3-layer MLP (Lin->ReLU->Lin->ReLU->Lin->LayerNorm) with WMMA bf16.
// mode 0: node_enc    rows=B*N_TOT  in=features(+zero h3)       -> dst0 = out ws
// mode 1: edge_enc    rows=E_ENC    in=enc_edge_attr            -> dst0 = ea ws
// mode 2: lat_e_enc   rows=E_LAT    in=lat_edge_attr            -> dst0 = lea (2 batch copies)
// mode 3: ep_mlp      rows=B*E_ENC  in=cat(out[src],out[dst],ea)-> residual+scatter into agg
// mode 4: np_mlp      rows=B*N_G    in=cat(out[mesh],agg)       -> residual, dst0 = d_out
// Weights for all 256-wide K chunks come pre-converted (bf16, [n][k]) and are
// staged by TDM; only the small layer-0 K (96/32) uses the f32 VALU path.
// ---------------------------------------------------------------------------
__global__ __launch_bounds__(THREADS)
void mlp3_ln_kernel(int mode, int rows, int K0pad, int K0valid,
                    const float* __restrict__ W0, const float* __restrict__ b0,
                    const float* __restrict__ b1, const float* __restrict__ b2,
                    const float* __restrict__ lng, const float* __restrict__ lnb,
                    const unsigned short* __restrict__ wt0,   // null for modes 0..2
                    const unsigned short* __restrict__ wt1,
                    const unsigned short* __restrict__ wt2,
                    const float* __restrict__ in0,
                    const float* __restrict__ outG,
                    const float* __restrict__ eaG,
                    const int*   __restrict__ encDst,
                    const int*   __restrict__ offs,
                    float* __restrict__ agg,
                    float* __restrict__ dst0) {
  extern __shared__ __align__(16) unsigned char smem[];
  unsigned short* actU = (unsigned short*)smem;       // [TM][AP] bf16
  unsigned short* wU   = actU + TM * AP;              // [256][WP] bf16 (Wt: [n][k])
  float*          outF = (float*)smem;                // [TM][256] f32, aliases act+W after barrier

  const int tid  = threadIdx.x;
  const int wv   = tid >> 5;
  const int lane = tid & 31;
  const int l16  = lane & 15;
  const int hi   = lane >> 4;
  const long row0 = (long)blockIdx.x * TM;

  int encOff = 0;
  if (mode == 3) encOff = offs[0] + 1;                // batched edge-index offset

  v8f acc[16];
  for (int l = 0; l < 3; ++l) {
    const int K  = (l == 0) ? K0pad : 256;
    const int Kv = (l == 0) ? K0valid : 256;

#pragma unroll
    for (int n = 0; n < 16; ++n)
#pragma unroll
      for (int i = 0; i < 8; ++i) acc[n][i] = 0.0f;

    for (int kc0 = 0; kc0 < K; kc0 += 256) {
      const int kc = (K - kc0 < 256) ? (K - kc0) : 256;
      __syncthreads();
      // ---- stage weights: TDM DMA for pre-converted 256-chunks ----
      const unsigned short* wsrc =
          (l == 2) ? wt2 : (l == 1) ? wt1
                         : (wt0 ? wt0 + (size_t)(kc0 >> 8) * CHUNK_ELEMS : nullptr);
      if (wsrc) {
#ifdef HAVE_TDM
        if (wv == 0) tdm_load_chunk(wsrc, (unsigned)(TM * AP * 2));
#else
        for (int t = tid; t < CHUNK_ELEMS; t += THREADS) {
          int n = t >> 8, k = t & 255;
          wU[n * WP + k] = wsrc[t];
        }
#endif
      } else {
        // small layer-0 K: convert f32 weights in VALU
        for (int t = tid; t < kc * 256; t += THREADS) {
          int k = t >> 8, n = t & 255;
          float w = (kc0 + k < Kv) ? W0[(size_t)(kc0 + k) * 256 + n] : 0.0f;
          wU[n * WP + k] = f2bf(w);
        }
      }
      // ---- stage layer-0 activation chunk (gather/concat per mode),
      //      overlapped with the in-flight TDM ----
      if (l == 0) {
        for (int t = tid; t < TM * kc; t += THREADS) {
          int r = t / kc, k = t - r * kc;
          int kg = kc0 + k;
          long gr = row0 + r;
          float v = 0.0f;
          if (gr < rows) {
            if (mode == 0) {
              int b = (int)(gr / N_TOT), i = (int)(gr - (long)b * N_TOT);
              if (i < N_LL && kg < F_IN) v = in0[((size_t)b * N_LL + i) * F_IN + kg];
            } else if (mode <= 2) {
              if (kg < 2) v = in0[(size_t)gr * 2 + kg];
            } else if (mode == 3) {
              int e = (int)gr, b = e >> 16, ee = e & (E_ENC - 1);
              int which = kg >> 8, kk = kg & 255;
              if (which == 0)      v = outG[(size_t)(ee + b * encOff) * 256 + kk];
              else if (which == 1) v = outG[(size_t)(encDst[ee] + b * encOff) * 256 + kk];
              else                 v = eaG[(size_t)ee * 256 + kk];
            } else {  // mode 4
              int r4 = (int)gr, b = r4 / N_G, i = r4 - b * N_G;
              int which = kg >> 8, kk = kg & 255;
              if (which == 0) v = outG[((size_t)(b * N_TOT + N_LL + i)) * 256 + kk];
              else            v = agg[(size_t)r4 * 256 + kk];
            }
          }
          actU[r * AP + k] = f2bf(v);
        }
      }
#ifdef HAVE_TDM
      if (wsrc && wv == 0) tdm_wait();
#endif
      __syncthreads();
      // ---- WMMA compute: 16 rows (this wave) x 256 cols, K in 32-steps ----
      const unsigned short* aRow = actU + (wv * 16 + l16) * AP;
      const int ksteps = kc >> 5;
      for (int ks = 0; ks < ksteps; ++ks) {
        const int k0 = ks << 5;
        v16bf a = ld_frag(aRow + k0 + hi * 8, aRow + k0 + hi * 8 + 16);
#pragma unroll
        for (int n = 0; n < 16; ++n) {
          const unsigned short* bRow = wU + (n * 16 + l16) * WP + k0 + hi * 8;
          v16bf bm = ld_frag(bRow, bRow + 16);
          acc[n] = __builtin_amdgcn_wmma_f32_16x16x32_bf16(
              false, a, false, bm, (short)0, acc[n], false, false);
        }
      }
    }

    if (l < 2) {  // bias + ReLU -> bf16 activations for next layer (wave-private rows)
      const float* bl = (l == 0) ? b0 : b1;
#pragma unroll
      for (int n = 0; n < 16; ++n) {
        int col = n * 16 + l16;
        float bv = bl[col];
#pragma unroll
        for (int v = 0; v < 8; ++v) {
          float x = acc[n][v] + bv;
          x = x > 0.0f ? x : 0.0f;
          actU[(wv * 16 + v + hi * 8) * AP + col] = f2bf(x);
        }
      }
    }
  }

  // ---- final: bias into f32 staging (aliases W/act LDS after barrier) ----
  __syncthreads();
#pragma unroll
  for (int n = 0; n < 16; ++n) {
    int col = n * 16 + l16;
    float bv = b2[col];
#pragma unroll
    for (int v = 0; v < 8; ++v)
      outF[(wv * 16 + v + hi * 8) * 256 + col] = acc[n][v] + bv;
  }
  __syncthreads();

  // ---- LayerNorm + mode-specific fused tail ----
  for (int r = 0; r < 16; ++r) {
    const int  row = wv * 16 + r;
    const long gr  = row0 + row;
    const float* x = outF + row * 256;
    float s = 0.0f, ss = 0.0f;
    for (int c = lane; c < 256; c += 32) { float v = x[c]; s += v; ss += v * v; }
#pragma unroll
    for (int m = 16; m >= 1; m >>= 1) { s += __shfl_xor(s, m, 32); ss += __shfl_xor(ss, m, 32); }
    if (gr >= rows) continue;
    const float mu  = s * (1.0f / 256.0f);
    const float var = ss * (1.0f / 256.0f) - mu * mu;
    const float inv = rsqrtf(var + 1e-5f);
    for (int c = lane; c < 256; c += 32) {
      float y = (x[c] - mu) * inv * lng[c] + lnb[c];
      if (mode == 0 || mode == 1) {
        dst0[gr * 256 + c] = y;
      } else if (mode == 2) {           // lea tiled over batch
        dst0[gr * 256 + c] = y;
        dst0[((long)E_LAT + gr) * 256 + c] = y;
      } else if (mode == 3) {           // residual + segment_sum scatter (mesh-compact agg)
        int e = (int)gr, b = e >> 16, ee = e & (E_ENC - 1);
        y += eaG[(size_t)ee * 256 + c];
        int dsc = encDst[ee] + b * encOff;
        int b2i = dsc / N_TOT, i2 = dsc - b2i * N_TOT;
        if (b2i < B_ && i2 >= N_LL)
          atomicAdd(agg + ((size_t)(b2i * N_G + (i2 - N_LL))) * 256 + c, y);
      } else {                          // mode 4: residual + final output
        int r4 = (int)gr, b = r4 / N_G, i = r4 - b * N_G;
        y += outG[((size_t)(b * N_TOT + N_LL + i)) * 256 + c];
        dst0[(long)r4 * 256 + c] = y;
      }
    }
  }
}

// zero agg + reset offset slots (ws is poisoned once, never re-poisoned)
__global__ void init_kernel(float* agg, long n, int* offs) {
  long i = (long)blockIdx.x * 256 + threadIdx.x;
  if (i == 0) { offs[0] = 0; offs[1] = 0; }
  for (; i < n; i += (long)gridDim.x * 256) agg[i] = 0.0f;
}

__global__ void max_kernel(const int* __restrict__ a, int n, int* slot) {
  int i = blockIdx.x * 256 + threadIdx.x;
  int m = 0;
  for (; i < n; i += gridDim.x * 256) m = a[i] > m ? a[i] : m;
#pragma unroll
  for (int s = 16; s >= 1; s >>= 1) { int o = __shfl_xor(m, s, 32); m = o > m ? o : m; }
  if ((threadIdx.x & 31) == 0) atomicMax(slot, m);
}

__global__ void lei_kernel(const int* __restrict__ lidx, const int* __restrict__ offs,
                           int* __restrict__ outLei) {
  int j = blockIdx.x * 256 + threadIdx.x;
  const int total = 2 * B_ * E_LAT;
  if (j >= total) return;
  int loff = offs[1] + 1;
  int dim = j / (B_ * E_LAT);
  int t   = j - dim * (B_ * E_LAT);
  int b   = t / E_LAT;
  int e   = t - b * E_LAT;
  outLei[j] = lidx[dim * E_LAT + e] + b * loff;
}

extern "C" void kernel_launch(void* const* d_in, const int* in_sizes, int n_in,
                              void* d_out, int out_size, void* d_ws, size_t ws_size,
                              hipStream_t stream) {
  (void)in_sizes; (void)n_in; (void)out_size; (void)ws_size;
  const float* features = (const float*)d_in[0];
  const int*   enc_ei   = (const int*)d_in[1];
  const float* enc_attr = (const float*)d_in[2];
  const int*   lat_ei   = (const int*)d_in[3];
  const float* lat_attr = (const float*)d_in[4];
  // jax dict pytree (sorted keys): W0,W1,W2,b0,b1,b2,ln_b,ln_g per MLP
  auto P = [&](int base, int j) { return (const float*)d_in[base + j]; };
  const int NE = 5, EE = 13, LE = 21, EP = 29, NP = 37;

  // workspace: out[B*N_TOT,256] | ea[E_ENC,256] | agg[B*N_G,256] | offsets | bf16 wt chunks
  float* ws   = (float*)d_ws;
  float* outG = ws;
  float* eaG  = outG + (size_t)B_ * N_TOT * 256;
  float* agg  = eaG + (size_t)E_ENC * 256;
  int*   offs = (int*)(agg + (size_t)B_ * N_G * 256);
  unsigned short* wt = (unsigned short*)(offs + 64);     // 15 chunks x 64K halves
  auto WT = [&](int c) { return wt + (size_t)c * CHUNK_ELEMS; };

  float* dNp  = (float*)d_out;                                        // [B*N_G,256]
  int*   dLei = (int*)((float*)d_out + (size_t)B_ * N_G * 256);       // [2,B*E_LAT]
  float* dLea = (float*)d_out + (size_t)B_ * N_G * 256 + (size_t)2 * B_ * E_LAT;

  init_kernel<<<2048, 256, 0, stream>>>(agg, (long)B_ * N_G * 256, offs);
  max_kernel<<<64, 256, 0, stream>>>(enc_ei, 2 * N_LL, offs + 0);
  max_kernel<<<64, 256, 0, stream>>>(lat_ei, 2 * E_LAT, offs + 1);

  // ---- pre-convert all 256-wide weight chunks to bf16 [n][k] for TDM ----
  struct WtJob { const float* W; int Kv; int kc0; int chunk; };
  const WtJob jobs[15] = {
      {P(NE,1), 256,   0,  0}, {P(NE,2), 256,   0,  1},   // node W1,W2
      {P(EE,1), 256,   0,  2}, {P(EE,2), 256,   0,  3},   // edge W1,W2
      {P(LE,1), 256,   0,  4}, {P(LE,2), 256,   0,  5},   // lat  W1,W2
      {P(EP,0), 768,   0,  6}, {P(EP,0), 768, 256,  7}, {P(EP,0), 768, 512, 8},  // ep W0
      {P(EP,1), 256,   0,  9}, {P(EP,2), 256,   0, 10},   // ep W1,W2
      {P(NP,0), 512,   0, 11}, {P(NP,0), 512, 256, 12},   // np W0
      {P(NP,1), 256,   0, 13}, {P(NP,2), 256,   0, 14},   // np W1,W2
  };
  for (int j = 0; j < 15; ++j)
    wtconv_kernel<<<256, 256, 0, stream>>>(jobs[j].W, jobs[j].Kv, jobs[j].kc0, WT(jobs[j].chunk));

  dim3 blk(THREADS);
  auto g = [](long rows) { return dim3((unsigned)((rows + TM - 1) / TM)); };

  // node_enc: x=[features; zeros] -> outG
  mlp3_ln_kernel<<<g((long)B_ * N_TOT), blk, SMEM_BYTES, stream>>>(
      0, B_ * N_TOT, 96, F_IN,
      P(NE,0), P(NE,3), P(NE,4), P(NE,5), P(NE,7), P(NE,6),
      nullptr, WT(0), WT(1),
      features, nullptr, nullptr, nullptr, offs, nullptr, outG);
  // edge_enc -> eaG
  mlp3_ln_kernel<<<g(E_ENC), blk, SMEM_BYTES, stream>>>(
      1, E_ENC, 32, 2,
      P(EE,0), P(EE,3), P(EE,4), P(EE,5), P(EE,7), P(EE,6),
      nullptr, WT(2), WT(3),
      enc_attr, nullptr, nullptr, nullptr, offs, nullptr, eaG);
  // lat_edge_enc -> d_out lea (both batch copies)
  mlp3_ln_kernel<<<g(E_LAT), blk, SMEM_BYTES, stream>>>(
      2, E_LAT, 32, 2,
      P(LE,0), P(LE,3), P(LE,4), P(LE,5), P(LE,7), P(LE,6),
      nullptr, WT(4), WT(5),
      lat_attr, nullptr, nullptr, nullptr, offs, nullptr, dLea);
  // ep_mlp: cat(out[src],out[dst],ea) -> +ea -> atomic scatter into agg
  mlp3_ln_kernel<<<g((long)B_ * E_ENC), blk, SMEM_BYTES, stream>>>(
      3, B_ * E_ENC, 768, 768,
      P(EP,0), P(EP,3), P(EP,4), P(EP,5), P(EP,7), P(EP,6),
      WT(6), WT(9), WT(10),
      nullptr, outG, eaG, enc_ei + N_LL, offs, agg, nullptr);
  // np_mlp on mesh rows only: cat(out,agg) -> +out -> d_out
  mlp3_ln_kernel<<<g((long)B_ * N_G), blk, SMEM_BYTES, stream>>>(
      4, B_ * N_G, 512, 512,
      P(NP,0), P(NP,3), P(NP,4), P(NP,5), P(NP,7), P(NP,6),
      WT(11), WT(13), WT(14),
      nullptr, outG, nullptr, nullptr, offs, agg, dNp);
  // batched latent edge index
  lei_kernel<<<(2 * B_ * E_LAT + 255) / 256, 256, 0, stream>>>(lat_ei, offs, dLei);
}
